// Decoder_9629316677808
// MI455X (gfx1250) — matrix operand order repaired and checked by reading, not compiled, not run on hardware
//
#include <hip/hip_runtime.h>
#include <math.h>

#define BB 16
#define NN 64
#define TT 32
#define DD 512
#define HH 256
#define KK 13
#define SS 2048   // N*T
#define G4 1024   // 4*H
#define H2 512    // 2*H

typedef __attribute__((ext_vector_type(16))) __bf16 v16bf;
typedef __attribute__((ext_vector_type(8)))  float  v8f;

union FragB { uint4 q[2]; v16bf v; };

__device__ __forceinline__ unsigned short f2bf(float f) {
  unsigned int u = __float_as_uint(f);
  unsigned int r = u + 0x7FFFu + ((u >> 16) & 1u);
  return (unsigned short)(r >> 16);
}
__device__ __forceinline__ float sigf(float x) { return 1.0f / (1.0f + __expf(-x)); }

// CDNA5 async memory->LDS copy (16B per lane), tracked by ASYNCcnt.
// %0 = per-lane LDS byte offset (wave-relative), %1 = per-lane 64-bit global address.
__device__ __forceinline__ void async_copy_b128(unsigned int lds_off, const void* gaddr) {
  asm volatile("global_load_async_to_lds_b128 %0, %1, off"
               :: "v"(lds_off), "v"(gaddr) : "memory");
}
__device__ __forceinline__ void wait_async0() {
  asm volatile("s_wait_asynccnt 0x0" ::: "memory");
}
// generic (flat) LDS pointer -> wave-relative LDS byte offset (addr[31:0])
__device__ __forceinline__ unsigned int lds_off32(const void* p) {
  return (unsigned int)(unsigned long long)p;
}

// ---------------------------------------------------------------------------
// Kernel 1: union/pack layer. One block per batch element.
// ---------------------------------------------------------------------------
__global__ __launch_bounds__(256)
void k_union(const float* __restrict__ x, const float* __restrict__ xgcn,
             const int* __restrict__ mask, const int* __restrict__ tags,
             unsigned short* __restrict__ newx, int* __restrict__ newtag,
             int* __restrict__ doclen, float* __restrict__ maskout)
{
  const int b = blockIdx.x;
  const int tid = threadIdx.x;           // 256 threads
  __shared__ int srcIdx[SS];             // dest -> source position
  __shared__ int psum[256];

  const int base = tid * 8;
  int run = 0;
  int mloc[8];
  #pragma unroll
  for (int k = 0; k < 8; ++k) { mloc[k] = mask[b * SS + base + k]; run += mloc[k]; }
  psum[tid] = run;
  __syncthreads();
  for (int off = 1; off < 256; off <<= 1) {
    int v = psum[tid];
    int add = (tid >= off) ? psum[tid - off] : 0;
    __syncthreads();
    psum[tid] = v + add;
    __syncthreads();
  }
  const int total = psum[255];
  int excl = psum[tid] - run;
  #pragma unroll
  for (int k = 0; k < 8; ++k) {
    if (mloc[k]) { srcIdx[excl] = base + k; ++excl; }
  }
  __syncthreads();
  if (tid == 0) doclen[b] = total;

  for (int r = tid; r < SS; r += 256) {
    const int valid = (r < total);
    newtag[b * SS + r]  = valid ? tags[b * SS + srcIdx[r]] : 0;
    maskout[b * SS + r] = valid ? 1.0f : 0.0f;
  }
  __syncthreads();

  for (int idx = tid; idx < SS * DD; idx += 256) {
    const int r = idx >> 9;          // / DD
    const int d = idx & (DD - 1);
    float v = xgcn[((size_t)b * NN + (r >> 5)) * DD + d];   // r/T, T=32
    if (r < total) v += x[((size_t)b * SS + srcIdx[r]) * DD + d];
    newx[((size_t)b * SS + r) * DD + d] = f2bf(v);
  }
}

// ---------------------------------------------------------------------------
// Kernel 2: f32 -> bf16 weight conversion.
// ---------------------------------------------------------------------------
__global__ void k_castw(const float* __restrict__ w, unsigned short* __restrict__ o, int n)
{
  int i = blockIdx.x * blockDim.x + threadIdx.x;
  if (i < n) o[i] = f2bf(w[i]);
}

// ---------------------------------------------------------------------------
// Kernel 3: input-gate GEMM.  C[32768 x 1024] = A[32768 x 512] * W^T,
// A,W bf16, C f32.  Block tile 128(M) x 64(N), 8 waves each 16x64.
// Double-buffered LDS staged with GLOBAL_LOAD_ASYNC_TO_LDS_B128 so the
// DMA of K-step k+1 overlaps the WMMA chain of K-step k; one barrier/step.
// ---------------------------------------------------------------------------
__global__ __launch_bounds__(256)
void k_gate_gemm(const unsigned short* __restrict__ A,
                 const unsigned short* __restrict__ W,
                 float* __restrict__ C)
{
  const int bm   = blockIdx.x * 128;
  const int bn   = blockIdx.y * 64;
  const int tid  = threadIdx.x;
  const int wid  = tid >> 5;
  const int lane = tid & 31;
  const int lm   = lane & 15;
  const int kh   = lane >> 4;

  __shared__ unsigned short As[2][128][40];   // padded rows (80B = 20 dwords)
  __shared__ unsigned short Bs[2][64][40];

  // per-thread staging chunk assignment (16B = 8 bf16 chunks)
  // A tile: 128 rows x 4 chunks = 512 chunks -> 2 per thread
  const int ar0 = tid >> 2,            aq0 = tid & 3;          // chunk tid
  const int ar1 = (tid + 256) >> 2,    aq1 = (tid + 256) & 3;  // chunk tid+256
  // B tile: 64 rows x 4 chunks = 256 chunks -> 1 per thread
  const int br  = tid >> 2,            bq  = tid & 3;

  const unsigned int asbase = lds_off32(&As[0][0][0]);
  const unsigned int bsbase = lds_off32(&Bs[0][0][0]);
  const unsigned int asz = 128 * 40 * 2;   // bytes per A buffer
  const unsigned int bsz = 64 * 40 * 2;    // bytes per B buffer

  auto stage = [&](int k0, int buf) {
    async_copy_b128(asbase + buf * asz + (ar0 * 40 + aq0 * 8) * 2,
                    A + ((size_t)(bm + ar0)) * DD + k0 + aq0 * 8);
    async_copy_b128(asbase + buf * asz + (ar1 * 40 + aq1 * 8) * 2,
                    A + ((size_t)(bm + ar1)) * DD + k0 + aq1 * 8);
    async_copy_b128(bsbase + buf * bsz + (br * 40 + bq * 8) * 2,
                    W + ((size_t)(bn + br)) * DD + k0 + bq * 8);
  };

  v8f acc[4] = {};

  stage(0, 0);
  wait_async0();
  __syncthreads();

  for (int kt = 0; kt < DD / 32; ++kt) {
    const int buf = kt & 1;
    if (kt + 1 < DD / 32) stage((kt + 1) * 32, buf ^ 1);

    // A fragment, 16-bit A layout: lane-half kh holds K chunks [kh*8,+8) and [16+kh*8,+8)
    FragB a;
    {
      const unsigned short* rowp = &As[buf][wid * 16 + lm][0];
      a.q[0] = *(const uint4*)(rowp + kh * 8);
      a.q[1] = *(const uint4*)(rowp + 16 + kh * 8);
    }
    #pragma unroll
    for (int nt = 0; nt < 4; ++nt) {
      // B fragment: column lm, K chunks [kh*8,+8) and [16+kh*8,+8)
      FragB bf;
      const unsigned short* rowp = &Bs[buf][nt * 16 + lm][0];
      bf.q[0] = *(const uint4*)(rowp + kh * 8);
      bf.q[1] = *(const uint4*)(rowp + 16 + kh * 8);
      acc[nt] = __builtin_amdgcn_wmma_f32_16x16x32_bf16(
          false, a.v, false, bf.v, (short)0, acc[nt], false, false);
    }

    wait_async0();       // next tile resident in LDS
    __syncthreads();     // everyone done reading current tile
  }

  // C/D layout: element (m,n): vgpr = m%8, lane = n + 16*(m/8)
  #pragma unroll
  for (int nt = 0; nt < 4; ++nt) {
    #pragma unroll
    for (int i = 0; i < 8; ++i) {
      const int m = bm + wid * 16 + i + 8 * kh;
      const int n = bn + nt * 16 + lm;
      C[(size_t)m * G4 + n] = acc[nt][i];
    }
  }
}

// ---------------------------------------------------------------------------
// Kernel 4: recurrent BiLSTM.  One persistent block per direction (512 thr =
// 16 waves).  Wave w owns hidden slice j0 = w*16 and computes its own
// i/f/g/o tiles (4 WMMA accumulators over K=256 in 8 steps), so the cell
// update stays entirely in the wave.  h is double-buffered bf16 in LDS,
// c lives in registers.  One barrier per timestep.
// ---------------------------------------------------------------------------
__global__ __launch_bounds__(512)
void k_lstm(const unsigned short* __restrict__ Whhf,
            const unsigned short* __restrict__ Whhb,
            const float* __restrict__ Gf,
            const float* __restrict__ Gb,
            const float* __restrict__ biasf,
            const float* __restrict__ biasb,
            const int* __restrict__ doclen,
            float* __restrict__ outbuf)
{
  const int dir = blockIdx.x;
  const unsigned short* Whh = dir ? Whhb : Whhf;
  const float* G    = dir ? Gb : Gf;
  const float* bias = dir ? biasb : biasf;

  const int tid  = threadIdx.x;
  const int wid  = tid >> 5;     // 0..15
  const int lane = tid & 31;
  const int lm   = lane & 15;
  const int kh   = lane >> 4;
  const int j0   = wid * 16;
  const int j    = j0 + lm;      // hidden unit index 0..255

  __shared__ unsigned short hbf[2][16][264];  // [buf][batch][H], 16B-aligned rows

  for (int idx = tid; idx < 16 * 264; idx += 512)
    ((unsigned short*)hbf[0])[idx] = 0;

  int Lb[8];
  #pragma unroll
  for (int i = 0; i < 8; ++i) Lb[i] = doclen[i + 8 * kh];

  float biasv[4];
  #pragma unroll
  for (int g = 0; g < 4; ++g) biasv[g] = bias[g * 256 + j];

  float c[8];
  #pragma unroll
  for (int i = 0; i < 8; ++i) c[i] = 0.0f;

  __syncthreads();

  for (int t = 0; t < SS; ++t) {
    const int rb = t & 1;
    const int wb = rb ^ 1;

    int pos[8];
    #pragma unroll
    for (int i = 0; i < 8; ++i) {
      const int L = Lb[i];
      pos[i] = (dir == 0) ? t : ((t < L) ? (L - 1 - t) : t);
    }

    // prefetch next timestep's x-gate rows (global_prefetch_b8)
    if (t + 1 < SS) {
      #pragma unroll
      for (int i = 0; i < 8; ++i) {
        const int L = Lb[i];
        const int tn = t + 1;
        const int pn = (dir == 0) ? tn : ((tn < L) ? (L - 1 - tn) : tn);
        __builtin_prefetch(&G[((size_t)(i + 8 * kh) * SS + pn) * G4 + j], 0, 3);
      }
    }

    // h_prev A-fragments from LDS (8 K-steps of 32)
    FragB a[8];
    #pragma unroll
    for (int kk = 0; kk < 8; ++kk) {
      const unsigned short* rowp = &hbf[rb][lm][kk * 32];
      a[kk].q[0] = *(const uint4*)(rowp + kh * 8);
      a[kk].q[1] = *(const uint4*)(rowp + 16 + kh * 8);
    }

    // init accumulators with precomputed x-gates + bias
    v8f acc[4];
    #pragma unroll
    for (int g = 0; g < 4; ++g) {
      const int n = g * 256 + j;
      #pragma unroll
      for (int i = 0; i < 8; ++i) {
        const int brow = i + 8 * kh;
        acc[g][i] = G[((size_t)brow * SS + pos[i]) * G4 + n] + biasv[g];
      }
    }

    // acc += h_prev @ Whh^T  (Whh rows are K-major, L2-resident)
    #pragma unroll
    for (int g = 0; g < 4; ++g) {
      const int n = g * 256 + j;
      #pragma unroll
      for (int kk = 0; kk < 8; ++kk) {
        FragB bf;
        const unsigned short* wp = Whh + (size_t)n * HH + kk * 32;
        bf.q[0] = *(const uint4*)(wp + kh * 8);
        bf.q[1] = *(const uint4*)(wp + 16 + kh * 8);
        acc[g] = __builtin_amdgcn_wmma_f32_16x16x32_bf16(
            false, a[kk].v, false, bf.v, (short)0, acc[g], false, false);
      }
    }

    // LSTM cell update; write h (f32 to global, bf16 to next LDS buffer)
    #pragma unroll
    for (int i = 0; i < 8; ++i) {
      const float iv = acc[0][i];
      const float fv = acc[1][i];
      const float gv = acc[2][i];
      const float ov = acc[3][i];
      const float cn = sigf(fv) * c[i] + sigf(iv) * tanhf(gv);
      c[i] = cn;
      const float h = sigf(ov) * tanhf(cn);
      const int brow = i + 8 * kh;
      hbf[wb][brow][j] = f2bf(h);
      outbuf[((size_t)brow * SS + pos[i]) * H2 + dir * HH + j] = h;
    }
    __syncthreads();
  }
}

// ---------------------------------------------------------------------------
// Kernel 5: logits = (out * mask) @ mlp_w^T + mlp_b.  K=13 is tiny; one
// thread per (row, k) output element.
// ---------------------------------------------------------------------------
__global__ void k_logits(const float* __restrict__ outbuf,
                         const float* __restrict__ mlp_w,
                         const float* __restrict__ mlp_b,
                         const int* __restrict__ doclen,
                         float* __restrict__ logits)
{
  const int idx = blockIdx.x * blockDim.x + threadIdx.x;
  if (idx >= BB * SS * KK) return;
  const int k    = idx % KK;
  const int row  = idx / KK;
  const int b    = row / SS;
  const int spos = row % SS;
  float v = mlp_b[k];
  if (spos < doclen[b]) {
    const float* orow = outbuf + (size_t)row * H2;
    const float* wrow = mlp_w + (size_t)k * H2;
    float acc = 0.0f;
    for (int d = 0; d < H2; d += 4) {
      const float4 a = *(const float4*)(orow + d);
      const float4 w = *(const float4*)(wrow + d);
      acc += a.x * w.x + a.y * w.y + a.z * w.z + a.w * w.w;
    }
    v += acc;
  }
  logits[idx] = v;
}

// ---------------------------------------------------------------------------
// Kernel 6: CRF log-likelihood.  One wave (32 threads) per batch element.
// ---------------------------------------------------------------------------
__global__ __launch_bounds__(32)
void k_crf(const float* __restrict__ logits,
           const int* __restrict__ newtag,
           const int* __restrict__ doclen,
           const float* __restrict__ trans,
           const float* __restrict__ startt,
           const float* __restrict__ endt,
           float* __restrict__ ll)
{
  const int b = blockIdx.x;
  const int lane = threadIdx.x;
  __shared__ float alpha[KK];
  __shared__ float tsh[KK * KK];
  __shared__ float red[32];
  const float* lg = logits + (size_t)b * SS * KK;
  const int* tg = newtag + (size_t)b * SS;
  const int L = doclen[b];

  for (int i = lane; i < KK * KK; i += 32) tsh[i] = trans[i];
  if (lane < KK) alpha[lane] = startt[lane] + lg[lane];
  __syncthreads();

  for (int t = 1; t < L; ++t) {
    float na = 0.0f;
    if (lane < KK) {
      float m = -1e30f;
      #pragma unroll
      for (int i = 0; i < KK; ++i) m = fmaxf(m, alpha[i] + tsh[i * KK + lane]);
      float s = 0.0f;
      #pragma unroll
      for (int i = 0; i < KK; ++i) s += __expf(alpha[i] + tsh[i * KK + lane] - m);
      na = m + __logf(s) + lg[t * KK + lane];
    }
    __syncthreads();
    if (lane < KK) alpha[lane] = na;
    __syncthreads();
  }

  float emit = 0.0f;
  for (int t = lane; t < L; t += 32) emit += lg[t * KK + tg[t]];
  float trs = 0.0f;
  for (int t = 1 + lane; t < L; t += 32) trs += tsh[tg[t - 1] * KK + tg[t]];
  red[lane] = emit + trs;
  __syncthreads();
  if (lane == 0) {
    float num = startt[tg[0]] + endt[tg[L - 1]];
    for (int i = 0; i < 32; ++i) num += red[i];
    float m = -1e30f;
    for (int i = 0; i < KK; ++i) m = fmaxf(m, alpha[i] + endt[i]);
    float s = 0.0f;
    for (int i = 0; i < KK; ++i) s += __expf(alpha[i] + endt[i] - m);
    ll[b] = num - (m + __logf(s));
  }
}

// ---------------------------------------------------------------------------
extern "C" void kernel_launch(void* const* d_in, const int* in_sizes, int n_in,
                              void* d_out, int out_size, void* d_ws, size_t ws_size,
                              hipStream_t stream)
{
  (void)in_sizes; (void)n_in; (void)out_size; (void)ws_size;
  const float* x      = (const float*)d_in[0];
  const float* xgcn   = (const float*)d_in[1];
  const int*   mask   = (const int*)d_in[2];
  const int*   tags   = (const int*)d_in[4];
  const float* w_ih_f = (const float*)d_in[5];
  const float* w_hh_f = (const float*)d_in[6];
  const float* b_f    = (const float*)d_in[7];
  const float* w_ih_b = (const float*)d_in[8];
  const float* w_hh_b = (const float*)d_in[9];
  const float* b_b    = (const float*)d_in[10];
  const float* mlp_w  = (const float*)d_in[11];
  const float* mlp_b  = (const float*)d_in[12];
  const float* trans  = (const float*)d_in[13];
  const float* startt = (const float*)d_in[14];
  const float* endt   = (const float*)d_in[15];

  // output layout: [logits B*S*K | new_mask B*S | ll B] (floats)
  float* logits  = (float*)d_out;
  float* maskout = logits + (size_t)BB * SS * KK;
  float* ll      = maskout + (size_t)BB * SS;

  // workspace carve-up
  char* ws = (char*)d_ws;
  size_t off = 0;
  auto ALLOC = [&](size_t bytes) -> void* {
    void* p = (void*)(ws + off);
    off += (bytes + 255) & ~(size_t)255;
    return p;
  };
  unsigned short* newx = (unsigned short*)ALLOC((size_t)BB * SS * DD * 2);
  unsigned short* wihf = (unsigned short*)ALLOC((size_t)G4 * DD * 2);
  unsigned short* wihb = (unsigned short*)ALLOC((size_t)G4 * DD * 2);
  unsigned short* whhf = (unsigned short*)ALLOC((size_t)G4 * HH * 2);
  unsigned short* whhb = (unsigned short*)ALLOC((size_t)G4 * HH * 2);
  float* Gf     = (float*)ALLOC((size_t)BB * SS * G4 * 4);
  float* Gb     = (float*)ALLOC((size_t)BB * SS * G4 * 4);
  float* outbuf = (float*)ALLOC((size_t)BB * SS * H2 * 4);
  int* newtag   = (int*)ALLOC((size_t)BB * SS * 4);
  int* dlen     = (int*)ALLOC(64);

  // 1) union/pack + bf16 convert
  k_union<<<BB, 256, 0, stream>>>(x, xgcn, mask, tags, newx, newtag, dlen, maskout);

  // 2) weight conversions
  k_castw<<<(G4 * DD + 255) / 256, 256, 0, stream>>>(w_ih_f, wihf, G4 * DD);
  k_castw<<<(G4 * DD + 255) / 256, 256, 0, stream>>>(w_ih_b, wihb, G4 * DD);
  k_castw<<<(G4 * HH + 255) / 256, 256, 0, stream>>>(w_hh_f, whhf, G4 * HH);
  k_castw<<<(G4 * HH + 255) / 256, 256, 0, stream>>>(w_hh_b, whhb, G4 * HH);

  // 3) bulk input-gate GEMMs (both directions)
  dim3 g2((BB * SS) / 128, G4 / 64);
  k_gate_gemm<<<g2, 256, 0, stream>>>(newx, wihf, Gf);
  k_gate_gemm<<<g2, 256, 0, stream>>>(newx, wihb, Gb);

  // 4) recurrent BiLSTM (2 persistent blocks, one per direction)
  k_lstm<<<2, 512, 0, stream>>>(whhf, whhb, Gf, Gb, b_f, b_b, dlen, outbuf);

  // 5) logits
  k_logits<<<(BB * SS * KK + 255) / 256, 256, 0, stream>>>(outbuf, mlp_w, mlp_b, dlen, logits);

  // 6) CRF log-likelihood
  k_crf<<<BB, 32, 0, stream>>>(logits, newtag, dlen, trans, startt, endt, ll);
}
